// LinkPredictionEncoder_16037407883983
// MI455X (gfx1250) — compile-verified, matrix operand order checked by reading, not run on hardware
//
#include <hip/hip_runtime.h>
#include <hip/hip_bf16.h>

#define GCN_N_NODES 100000
#define GCN_N_EDGES 1600000

typedef __attribute__((ext_vector_type(16))) __bf16 v16bf;
typedef __attribute__((ext_vector_type(8)))  float  v8f;

// ---------------------------------------------------------------------------
// CDNA5 async global->LDS copy (ASYNCcnt), per CDNA5 ISA.
// GVS mode: mem = SADDR(64) + VADDR(32) ; VDST holds the LDS byte address.
// ---------------------------------------------------------------------------
__device__ __forceinline__ void async_g2l_b128(uint32_t lds_addr, uint32_t goff,
                                               const void* base) {
  asm volatile("global_load_async_to_lds_b128 %0, %1, %2"
               :: "v"(lds_addr), "v"(goff), "s"(base)
               : "memory");
}
__device__ __forceinline__ void wait_asynccnt0() {
  asm volatile("s_wait_asynccnt 0" ::: "memory");
}

// ---------------------------------------------------------------------------
// zero fill
// ---------------------------------------------------------------------------
__global__ __launch_bounds__(256) void zero_kernel(float* __restrict__ p, int n) {
  int i = blockIdx.x * 256 + threadIdx.x;
  int stride = gridDim.x * 256;
  for (; i < n; i += stride) p[i] = 0.0f;
}

// ---------------------------------------------------------------------------
// one-time weight prep: Wt[n][k] = (bf16) W[k][n]   (K = 128 always)
// bf16 + pre-transposed == exactly the B-fragment layout; enables pure
// async DMA staging of B in the GEMM (no per-block convert/transpose).
// ---------------------------------------------------------------------------
__global__ __launch_bounds__(256) void convert_w_kernel(
    const float* __restrict__ W, __bf16* __restrict__ Wt, int N) {
  int idx = blockIdx.x * 256 + threadIdx.x;
  if (idx >= N * 128) return;
  int n = idx >> 7;
  int k = idx & 127;
  Wt[idx] = (__bf16)W[k * N + n];
}

// ---------------------------------------------------------------------------
// edge aggregation: one wave32 per edge; lane handles 4 strided floats.
// agg[row] += h[col]; deg[row] += 1. L2-resident (h = 51MB < 192MB L2).
// ---------------------------------------------------------------------------
__global__ __launch_bounds__(256) void aggregate_kernel(
    const float* __restrict__ h,
    const int*   __restrict__ erow,
    const int*   __restrict__ ecol,
    float* __restrict__ agg,
    float* __restrict__ degv,
    int E)
{
  int gid  = blockIdx.x * 256 + threadIdx.x;
  int wid  = gid >> 5;
  int lane = gid & 31;
  if (wid >= E) return;
  int r = erow[wid];
  int c = ecol[wid];
  const float* src = h   + (size_t)c * 128;
  float*       dst = agg + (size_t)r * 128;
#pragma unroll
  for (int q = 0; q < 4; ++q) {
    float v = src[q * 32 + lane];
    unsafeAtomicAdd(&dst[q * 32 + lane], v);
  }
  if (lane == 0) unsafeAtomicAdd(&degv[r], 1.0f);
}

// ---------------------------------------------------------------------------
// WMMA GEMM:  out[M][NDIM] = A[M][128] @ W[128][NDIM] + bias
//   B comes in pre-transposed bf16 (Wt[n][k]) and is staged to LDS via
//   global_load_async_to_lds_b128 (ASYNCcnt DMA).
// MODE 0: plain bias store (input projection / output head)
// MODE 1: A scaled by 1/max(deg,1); fused bias + residual + LayerNorm + ReLU,
//         written in place to Hio (block owns its 64 rows -> race free)
// Block = 256 threads = 8 wave32. 64-row M tile. Full K staged in LDS (bf16).
// ---------------------------------------------------------------------------
template<int NDIM, int MODE>
__global__ __launch_bounds__(256) void gemm_kernel(
    const float*  __restrict__ A,      // [M][128] fp32
    const __bf16* __restrict__ Wt,     // [NDIM][128] bf16 (transposed, row = 256B)
    const float*  __restrict__ bias,   // [NDIM]
    float*        Hio,                 // MODE0: out [M][NDIM]; MODE1: residual-in + out [M][128]
    const float*  __restrict__ degv,   // MODE1 only
    const float*  __restrict__ gamma,  // MODE1 only
    const float*  __restrict__ beta,   // MODE1 only
    int M)
{
  constexpr int NT = NDIM / 32;                 // n-tiles (16 cols each) per wave
  __shared__ __bf16 As[64][136];                // padded stride vs bank conflicts
  __shared__ __bf16 Bs[NDIM][136];              // [n][k], padded

  const int tid   = threadIdx.x;
  const int wave  = tid >> 5;
  const int lane  = tid & 31;
  const int m0blk = blockIdx.x * 64;

  // ---- stage B via async DMA: NDIM rows x 256B, 16B chunks, padded LDS rows
  {
    const uint32_t bsbase = (uint32_t)(uintptr_t)(&Bs[0][0]);
#pragma unroll
    for (int i = 0; i < NDIM / 16; ++i) {        // NDIM*16 chunks total / 256 thr
      int idx = tid + i * 256;
      int n   = idx >> 4;                        // weight row (N index)
      int c   = idx & 15;                        // 16B chunk within row
      uint32_t goff = (uint32_t)(n * 256 + c * 16);
      uint32_t loff = bsbase + (uint32_t)(n * 272 + c * 16);  // 136*2B stride
      async_g2l_b128(loff, goff, (const void*)Wt);
    }
  }

  // ---- stage A tile (64x128 fp32 -> bf16), scaled by 1/deg in MODE 1 ----
#pragma unroll
  for (int i = 0; i < 8; ++i) {
    int idx  = tid + i * 256;        // float4 index, 0..2047
    int rl   = idx >> 5;             // local row 0..63
    int c4   = idx & 31;             // float4 column
    int grow = m0blk + rl;
    float4 v = make_float4(0.f, 0.f, 0.f, 0.f);
    float  sc = 1.0f;
    if (grow < M) {
      v = *reinterpret_cast<const float4*>(A + (size_t)grow * 128 + c4 * 4);
      if (MODE == 1) sc = 1.0f / fmaxf(degv[grow], 1.0f);
    }
    As[rl][c4 * 4 + 0] = (__bf16)(v.x * sc);
    As[rl][c4 * 4 + 1] = (__bf16)(v.y * sc);
    As[rl][c4 * 4 + 2] = (__bf16)(v.z * sc);
    As[rl][c4 * 4 + 3] = (__bf16)(v.w * sc);
  }

  wait_asynccnt0();     // our wave's async B chunks are in LDS
  __syncthreads();      // + everyone else's (and the ds_store'd A tile)

  // ---- WMMA main loop ----
  const int mt  = wave & 3;                     // M sub-tile (16 rows)
  const int n0w = (wave >> 2) * (NT * 16);      // N half
  const int mr  = lane & 15;
  const int khi = (lane >> 4) << 3;             // lanes 16-31 hold K+8 block

  v8f acc[NT];
#pragma unroll
  for (int t = 0; t < NT; ++t)
#pragma unroll
    for (int j = 0; j < 8; ++j) acc[t][j] = 0.0f;

#pragma unroll
  for (int kk = 0; kk < 128; kk += 32) {
    // A fragment: lane = M row; elems 0-7 -> K kk+khi+0..7, 8-15 -> kk+khi+16..23
    v16bf a;
    const __bf16* ap = &As[mt * 16 + mr][kk + khi];
#pragma unroll
    for (int j = 0; j < 8; ++j) { a[j] = ap[j]; a[8 + j] = ap[16 + j]; }
#pragma unroll
    for (int t = 0; t < NT; ++t) {
      v16bf b;  // B fragment: lane = N col (Bs transposed), same K pattern
      const __bf16* bp = &Bs[n0w + t * 16 + mr][kk + khi];
#pragma unroll
      for (int j = 0; j < 8; ++j) { b[j] = bp[j]; b[8 + j] = bp[16 + j]; }
      acc[t] = __builtin_amdgcn_wmma_f32_16x16x32_bf16(
          /*neg_a=*/false, a, /*neg_b=*/false, b,
          /*c_mod=*/(short)0, acc[t], /*reuse_a=*/false, /*reuse_b=*/false);
    }
  }

  // C layout: VGPR r, lanes 0-15 -> M=r, lanes 16-31 -> M=8+r; N = lane&15
  const int roff = (lane >> 4) << 3;

  if (MODE == 0) {
#pragma unroll
    for (int t = 0; t < NT; ++t) {
      int n = n0w + t * 16 + mr;
      float bn = bias[n];
#pragma unroll
      for (int r = 0; r < 8; ++r) {
        int grow = m0blk + mt * 16 + roff + r;
        if (grow < M) Hio[(size_t)grow * NDIM + n] = acc[t][r] + bn;
      }
    }
  } else {
    __shared__ float Cs[64 * 128];
    // bias + residual -> LDS tile
#pragma unroll
    for (int t = 0; t < NT; ++t) {
      int n = n0w + t * 16 + mr;
      float bn = bias[n];
#pragma unroll
      for (int r = 0; r < 8; ++r) {
        int ml   = mt * 16 + roff + r;
        int grow = m0blk + ml;
        float res = (grow < M) ? Hio[(size_t)grow * 128 + n] : 0.0f;
        Cs[ml * 128 + n] = acc[t][r] + bn + res;
      }
    }
    __syncthreads();

    // LayerNorm + ReLU, one wave per 8 rows; wave32 shuffle reduction over 128
#pragma unroll
    for (int rr = 0; rr < 8; ++rr) {
      int ml   = wave * 8 + rr;
      int grow = m0blk + ml;
      float v0[4];
      float s = 0.0f, s2 = 0.0f;
#pragma unroll
      for (int q = 0; q < 4; ++q) {
        float v = Cs[ml * 128 + lane * 4 + q];
        v0[q] = v; s += v; s2 += v * v;
      }
#pragma unroll
      for (int off = 16; off >= 1; off >>= 1) {
        s  += __shfl_xor(s,  off, 32);
        s2 += __shfl_xor(s2, off, 32);
      }
      float mu  = s  * (1.0f / 128.0f);
      float var = s2 * (1.0f / 128.0f) - mu * mu;
      float rs  = rsqrtf(var + 1e-5f);
      if (grow < M) {
#pragma unroll
        for (int q = 0; q < 4; ++q) {
          int n = lane * 4 + q;
          float y = (v0[q] - mu) * rs * gamma[n] + beta[n];
          Hio[(size_t)grow * 128 + n] = fmaxf(y, 0.0f);
        }
      }
    }
  }
}

// ---------------------------------------------------------------------------
// launch: prep bf16 weights; h = x@w_in+b_in;
//         2x { zero; aggregate; conv+LN+ReLU }; out = h@w_out+b_out
// ---------------------------------------------------------------------------
extern "C" void kernel_launch(void* const* d_in, const int* in_sizes, int n_in,
                              void* d_out, int out_size, void* d_ws, size_t ws_size,
                              hipStream_t stream) {
  const float* x     = (const float*)d_in[0];
  const int*   ei    = (const int*)  d_in[1];
  const float* w_in  = (const float*)d_in[2];
  const float* b_in  = (const float*)d_in[3];
  const float* w_c0  = (const float*)d_in[4];
  const float* b_c0  = (const float*)d_in[5];
  const float* w_c1  = (const float*)d_in[6];
  const float* b_c1  = (const float*)d_in[7];
  const float* g0    = (const float*)d_in[8];
  const float* be0   = (const float*)d_in[9];
  const float* g1    = (const float*)d_in[10];
  const float* be1   = (const float*)d_in[11];
  const float* w_out = (const float*)d_in[12];
  const float* b_out = (const float*)d_in[13];
  float* out = (float*)d_out;

  const int M = GCN_N_NODES, E = GCN_N_EDGES;

  // workspace layout (16B-aligned sections)
  float* h   = (float*)d_ws;                 // [M][128] fp32
  float* agg = h   + (size_t)M * 128;        // [M][128] fp32
  float* deg = agg + (size_t)M * 128;        // [M] fp32 (contiguous after agg)
  __bf16* wt_in  = (__bf16*)(deg + M);       // [128][128] bf16 transposed
  __bf16* wt_c0  = wt_in  + 128 * 128;
  __bf16* wt_c1  = wt_c0  + 128 * 128;
  __bf16* wt_out = wt_c1  + 128 * 128;       // [64][128] bf16 transposed
  const int* erow = ei;
  const int* ecol = ei + E;

  dim3 blk(256);
  int gemmBlocks = (M + 63) / 64;
  int aggBlocks  = (E * 32 + 255) / 256;     // one wave per edge

  // one-time bf16 transposed weight prep (tiny, L2-resident thereafter)
  convert_w_kernel<<<(128 * 128 + 255) / 256, blk, 0, stream>>>(w_in, wt_in, 128);
  convert_w_kernel<<<(128 * 128 + 255) / 256, blk, 0, stream>>>(w_c0, wt_c0, 128);
  convert_w_kernel<<<(128 * 128 + 255) / 256, blk, 0, stream>>>(w_c1, wt_c1, 128);
  convert_w_kernel<<<(64  * 128 + 255) / 256, blk, 0, stream>>>(w_out, wt_out, 64);

  // input projection
  gemm_kernel<128, 0><<<gemmBlocks, blk, 0, stream>>>(
      x, wt_in, b_in, h, nullptr, nullptr, nullptr, M);

  const __bf16* wc[2] = {wt_c0, wt_c1};
  const float*  bc[2] = {b_c0, b_c1};
  const float*  gg[2] = {g0, g1};
  const float*  bb[2] = {be0, be1};

  for (int l = 0; l < 2; ++l) {
    zero_kernel<<<2048, blk, 0, stream>>>(agg, M * 128 + M);
    aggregate_kernel<<<aggBlocks, blk, 0, stream>>>(h, erow, ecol, agg, deg, E);
    gemm_kernel<128, 1><<<gemmBlocks, blk, 0, stream>>>(
        agg, wc[l], bc[l], h, deg, gg[l], bb[l], M);
  }

  // output head
  gemm_kernel<64, 0><<<gemmBlocks, blk, 0, stream>>>(
      h, wt_out, b_out, out, nullptr, nullptr, nullptr, M);
}
